// EdgeDegreeEmbedding_30743375905281
// MI455X (gfx1250) — matrix operand order repaired and checked by reading, not compiled, not run on hardware
//
#include <hip/hip_runtime.h>
#include <hip/hip_fp16.h>

typedef __attribute__((ext_vector_type(16))) _Float16 v16h;
typedef __attribute__((ext_vector_type(8)))  float    v8f;
typedef __attribute__((ext_vector_type(4)))  float    v4f;

#define N_NODES   12500
#define N_EDGES   400000
#define N_RADIAL  128
#define HIDDEN    64
#define SPHERE_C  64
#define M0        5
#define M_ALL     19
#define FULL      25
#define IN_DIM    256          // 128 + 64 + 64
#define OUT_DIM   320          // M0 * SPHERE_C
#define RESCALE   5.0f

#define EPB       32           // edges per block  (2 M-tiles of 16)
#define THREADS   256          // 8 wave32s
#define A_STRIDE  264          // 256 + 8 halves  -> bank-conflict-free b128 lane loads
#define H_STRIDE  72           // 64 + 8 halves
#define C_STRIDE  68           // 64 + 4 floats
#define H3_STRIDE 328          // 320 + 8 halves
#define WIG_ROWS  (EPB * FULL) // 800 rows of 5 useful floats
#define WIG_STRIDE 6           // floats per row in LDS (24 B, 8B-aligned for b64)

// d_ws layout (f16): W1t[64][256] @0, W2t[64][64] @16384, W3t[320][64] @20480
#define WS_W1T 0
#define WS_W2T 16384
#define WS_W3T 20480

struct SMem {
    union {
        _Float16 A[EPB * A_STRIDE];    // phase 1 input  (16896 B)
        _Float16 H3[EPB * H3_STRIDE];  // phase 3 output (20992 B)
    } u;
    _Float16 H1[EPB * H_STRIDE];       // 4608 B
    _Float16 H2[EPB * H_STRIDE];       // 4608 B
    float    Csc[EPB * C_STRIDE];      // 8704 B, f32 GEMM staging for LayerNorm
    float    WIG[WIG_ROWS * WIG_STRIDE]; // 19200 B, async-staged wigner[:, :, :5]
    int      spcS[EPB];
    int      spcR[EPB];
    float    env[EPB];
    int      recv[EPB];
};                                      // total ~58.9 KB

__device__ __forceinline__ float wave_sum(float v) {
    #pragma unroll
    for (int off = 16; off > 0; off >>= 1) v += __shfl_xor(v, off, 32);
    return v;
}

// Async global -> LDS copies (CDNA5, tracked with ASYNCcnt).
__device__ __forceinline__ void async_g2l_b64(unsigned lds_off, const float* gaddr) {
    asm volatile("global_load_async_to_lds_b64 %0, %1, off"
                 :: "v"(lds_off), "v"(gaddr) : "memory");
}
__device__ __forceinline__ void async_g2l_b32(unsigned lds_off, const float* gaddr) {
    asm volatile("global_load_async_to_lds_b32 %0, %1, off"
                 :: "v"(lds_off), "v"(gaddr) : "memory");
}
__device__ __forceinline__ void wait_asynccnt0() {
    asm volatile("s_wait_asynccnt 0" ::: "memory");
}

// A-fragment (16x32 f16) from LDS, row-major with padded stride.
// Lane l in [0,16): row M=l, K = kb..kb+7 and kb+16..kb+23 ; lanes 16..31: +8 K offset.
__device__ __forceinline__ v16h load_a_frag(const _Float16* base, int stride,
                                            int mt, int kt, int low, int hs) {
    const _Float16* p = base + (mt * 16 + low) * stride + kt * 32 + hs * 8;
    union { v16h h; v4f f[2]; } u;
    u.f[0] = *(const v4f*)(p);
    u.f[1] = *(const v4f*)(p + 16);
    return u.h;
}

// B-fragment (32x16 f16) from transposed weights Wt[n][k] in global (L2-resident).
// Lane l in [0,16): col N=l, K = kb..kb+7 and kb+16..kb+23 ; lanes 16..31: +8 K offset.
__device__ __forceinline__ v16h load_b_frag(const _Float16* wt, int stride,
                                            int nt, int kt, int low, int hs) {
    const _Float16* p = wt + (nt * 16 + low) * stride + kt * 32 + hs * 8;
    union { v16h h; v4f f[2]; } u;
    u.f[0] = *(const v4f*)(p);
    u.f[1] = *(const v4f*)(p + 16);
    return u.h;
}

__device__ __forceinline__ v8f wmma_f16(v16h a, v16h b, v8f c) {
    return __builtin_amdgcn_wmma_f32_16x16x32_f16(false, a, false, b,
                                                  (short)0, c, false, false);
}

// C tile (16x16 f32, 8 VGPRs) -> LDS f32 scratch
__device__ __forceinline__ void store_c(float* Csc, const v8f& c,
                                        int mt, int nt, int low, int hs) {
    int col = nt * 16 + low;
    #pragma unroll
    for (int j = 0; j < 8; ++j)
        Csc[(mt * 16 + hs * 8 + j) * C_STRIDE + col] = c[j];
}

// LayerNorm(x + b) * g + be, then SiLU; Csc[32][64] f32 -> H[32][64] f16
__device__ __forceinline__ void ln_silu(const float* Csc, _Float16* H,
                                        const float* b, const float* g,
                                        const float* be, int wave, int lane) {
    #pragma unroll
    for (int j = 0; j < 4; ++j) {
        int r = wave * 4 + j;
        float x0 = Csc[r * C_STRIDE + lane]      + b[lane];
        float x1 = Csc[r * C_STRIDE + lane + 32] + b[lane + 32];
        float mu  = wave_sum(x0 + x1) * (1.0f / 64.0f);
        float d0 = x0 - mu, d1 = x1 - mu;
        float var = wave_sum(d0 * d0 + d1 * d1) * (1.0f / 64.0f);
        float inv = rsqrtf(var + 1e-5f);
        float y0 = d0 * inv * g[lane]      + be[lane];
        float y1 = d1 * inv * g[lane + 32] + be[lane + 32];
        y0 = y0 * (1.0f / (1.0f + __expf(-y0)));
        y1 = y1 * (1.0f / (1.0f + __expf(-y1)));
        H[r * H_STRIDE + lane]      = (_Float16)y0;
        H[r * H_STRIDE + lane + 32] = (_Float16)y1;
    }
}

__global__ void prep_weights_kernel(const float* __restrict__ W1,
                                    const float* __restrict__ W2,
                                    const float* __restrict__ W3,
                                    _Float16* __restrict__ ws) {
    const int total = 256 * 64 + 64 * 64 + 64 * 320;
    for (int i = blockIdx.x * blockDim.x + threadIdx.x; i < total;
         i += gridDim.x * blockDim.x) {
        if (i < 16384) {                 // W1t[n][k] = W1[k][n], 256x64
            int n = i >> 8, k = i & 255;
            ws[WS_W1T + n * 256 + k] = (_Float16)W1[k * 64 + n];
        } else if (i < 20480) {          // W2t[n][k], 64x64
            int j = i - 16384; int n = j >> 6, k = j & 63;
            ws[WS_W2T + n * 64 + k] = (_Float16)W2[k * 64 + n];
        } else {                         // W3t[n][k] = W3[k][n], 64x320
            int j = i - 20480; int n = j >> 6, k = j & 63;
            ws[WS_W3T + n * 64 + k] = (_Float16)W3[k * 320 + n];
        }
    }
}

__global__ void zero_out_kernel(float* __restrict__ out, int n4) {
    v4f z = {0.f, 0.f, 0.f, 0.f};
    for (int i = blockIdx.x * blockDim.x + threadIdx.x; i < n4;
         i += gridDim.x * blockDim.x)
        ((v4f*)out)[i] = z;
}

__global__ __launch_bounds__(THREADS)
void edge_degree_fused_kernel(const int*   __restrict__ node_species,
                              const float* __restrict__ dist,
                              const int*   __restrict__ senders,
                              const int*   __restrict__ receivers,
                              const float* __restrict__ wigner,
                              const float* __restrict__ envelope,
                              const float* __restrict__ send_emb,
                              const float* __restrict__ recv_emb,
                              const float* __restrict__ b1,
                              const float* __restrict__ g1,
                              const float* __restrict__ be1,
                              const float* __restrict__ b2,
                              const float* __restrict__ g2,
                              const float* __restrict__ be2,
                              const float* __restrict__ b3,
                              const _Float16* __restrict__ ws,
                              float* __restrict__ out) {
    __shared__ SMem sm;
    const int tid  = threadIdx.x;
    const int wave = tid >> 5;
    const int lane = tid & 31;
    const int low  = lane & 15;
    const int hs   = lane >> 4;
    const int e0   = blockIdx.x * EPB;

    // ---- phase 0a: per-edge metadata ----
    if (tid < EPB) {
        int s = senders[e0 + tid];
        sm.spcS[tid] = node_species[s];
    } else if (tid < 2 * EPB) {
        int t = tid - EPB;
        int r = receivers[e0 + t];
        sm.spcR[t] = node_species[r];
        sm.recv[t] = r;
    } else if (tid < 3 * EPB) {
        int t = tid - 2 * EPB;
        sm.env[t] = envelope[e0 + t];
    }

    // ---- phase 0b: kick off ASYNC global->LDS staging of wigner[:, :, :5]
    //      (5 floats of every 19-float row; 800 rows, 24B LDS stride).
    //      These overlap all three GEMM phases; tracked by ASYNCcnt. ----
    {
        const unsigned wig_base = (unsigned)(size_t)(&sm.WIG[0]);
        for (int r = tid; r < WIG_ROWS; r += THREADS) {
            int e = r / FULL, f = r - e * FULL;
            const float* g = wigner + (size_t)(e0 + e) * (FULL * M_ALL) + f * M_ALL;
            unsigned l = wig_base + (unsigned)(r * (WIG_STRIDE * 4));
            async_g2l_b64(l,      g);      // m = 0,1
            async_g2l_b64(l + 8,  g + 2);  // m = 2,3
            async_g2l_b32(l + 16, g + 4);  // m = 4
        }
    }
    __syncthreads();

    // ---- phase 0c: build A = [dist | s_emb | r_emb] as f16 ----
    for (int i = tid; i < EPB * IN_DIM; i += THREADS) {
        int e = i >> 8, k = i & 255;
        float v;
        if (k < N_RADIAL)       v = dist[(size_t)(e0 + e) * N_RADIAL + k];
        else if (k < 192)       v = send_emb[sm.spcS[e] * HIDDEN + (k - 128)];
        else                    v = recv_emb[sm.spcR[e] * HIDDEN + (k - 192)];
        sm.u.A[e * A_STRIDE + k] = (_Float16)v;
    }
    __syncthreads();

    const int mt = wave & 1;       // M-tile (2 tiles of 16 edges)

    // ---- phase 1: GEMM1  [32x256] @ [256x64] (f16 WMMA, f32 acc) ----
    {
        const int nt = wave >> 1;  // 4 N-tiles
        v8f acc = {0.f, 0.f, 0.f, 0.f, 0.f, 0.f, 0.f, 0.f};
        #pragma unroll
        for (int kt = 0; kt < 8; ++kt) {
            v16h a = load_a_frag(sm.u.A, A_STRIDE, mt, kt, low, hs);
            v16h b = load_b_frag(ws + WS_W1T, 256, nt, kt, low, hs);
            acc = wmma_f16(a, b, acc);
        }
        store_c(sm.Csc, acc, mt, nt, low, hs);
    }
    __syncthreads();
    ln_silu(sm.Csc, sm.H1, b1, g1, be1, wave, lane);
    __syncthreads();

    // ---- phase 2: GEMM2  [32x64] @ [64x64] ----
    {
        const int nt = wave >> 1;
        v8f acc = {0.f, 0.f, 0.f, 0.f, 0.f, 0.f, 0.f, 0.f};
        #pragma unroll
        for (int kt = 0; kt < 2; ++kt) {
            v16h a = load_a_frag(sm.H1, H_STRIDE, mt, kt, low, hs);
            v16h b = load_b_frag(ws + WS_W2T, 64, nt, kt, low, hs);
            acc = wmma_f16(a, b, acc);
        }
        store_c(sm.Csc, acc, mt, nt, low, hs);
    }
    __syncthreads();
    ln_silu(sm.Csc, sm.H2, b2, g2, be2, wave, lane);
    __syncthreads();

    // ---- phase 3: GEMM3  [32x64] @ [64x320] -> h3 f16 (overlays A) ----
    {
        v16h a0 = load_a_frag(sm.H2, H_STRIDE, mt, 0, low, hs);
        v16h a1 = load_a_frag(sm.H2, H_STRIDE, mt, 1, low, hs);
        #pragma unroll
        for (int j = 0; j < 5; ++j) {
            const int nt = (wave >> 1) + 4 * j;   // covers 0..19
            v8f acc = {0.f, 0.f, 0.f, 0.f, 0.f, 0.f, 0.f, 0.f};
            acc = wmma_f16(a0, load_b_frag(ws + WS_W3T, 64, nt, 0, low, hs), acc);
            acc = wmma_f16(a1, load_b_frag(ws + WS_W3T, 64, nt, 1, low, hs), acc);
            const int col = nt * 16 + low;
            const float bb = b3[col];
            #pragma unroll
            for (int jj = 0; jj < 8; ++jj) {
                int row = mt * 16 + hs * 8 + jj;
                sm.u.H3[row * H3_STRIDE + col] = (_Float16)(acc[jj] + bb);
            }
        }
    }
    // every wave drains its own ASYNCcnt, then the workgroup barrier makes
    // all async LDS writes (and GEMM3 h3 stores) visible to everyone
    wait_asynccnt0();
    __syncthreads();

    // ---- phase 4: Wigner contraction [25x5]@[5x64], envelope, scatter-add ----
    // wave handles 4 edges; lane covers channels (lane, lane+32);
    // wigner coefficients come from LDS (broadcast reads, staged async)
    #pragma unroll
    for (int jj = 0; jj < 4; ++jj) {
        const int e = wave * 4 + jj;
        float f0[M0], f1[M0];
        #pragma unroll
        for (int m = 0; m < M0; ++m) {
            f0[m] = (float)sm.u.H3[e * H3_STRIDE + m * SPHERE_C + lane];
            f1[m] = (float)sm.u.H3[e * H3_STRIDE + m * SPHERE_C + lane + 32];
        }
        const float scale = sm.env[e] * (1.0f / RESCALE);
        float* obase = out + (size_t)sm.recv[e] * (FULL * SPHERE_C);
        const float* wrow = &sm.WIG[e * FULL * WIG_STRIDE];
        #pragma unroll
        for (int f = 0; f < FULL; ++f) {
            float a0 = 0.f, a1 = 0.f;
            #pragma unroll
            for (int m = 0; m < M0; ++m) {
                float w = wrow[f * WIG_STRIDE + m];   // lane-uniform LDS broadcast
                a0 = fmaf(w, f0[m], a0);
                a1 = fmaf(w, f1[m], a1);
            }
            unsafeAtomicAdd(obase + f * SPHERE_C + lane,      a0 * scale);
            unsafeAtomicAdd(obase + f * SPHERE_C + lane + 32, a1 * scale);
        }
    }
}

extern "C" void kernel_launch(void* const* d_in, const int* in_sizes, int n_in,
                              void* d_out, int out_size, void* d_ws, size_t ws_size,
                              hipStream_t stream) {
    const int*   node_species = (const int*)  d_in[0];
    const float* dist         = (const float*)d_in[1];
    const int*   senders      = (const int*)  d_in[2];
    const int*   receivers    = (const int*)  d_in[3];
    const float* wigner       = (const float*)d_in[4];
    const float* envelope     = (const float*)d_in[5];
    const float* send_emb     = (const float*)d_in[6];
    const float* recv_emb     = (const float*)d_in[7];
    const float* W1 = (const float*)d_in[8];
    const float* b1 = (const float*)d_in[9];
    const float* g1 = (const float*)d_in[10];
    const float* be1= (const float*)d_in[11];
    const float* W2 = (const float*)d_in[12];
    const float* b2 = (const float*)d_in[13];
    const float* g2 = (const float*)d_in[14];
    const float* be2= (const float*)d_in[15];
    const float* W3 = (const float*)d_in[16];
    const float* b3 = (const float*)d_in[17];

    float*    out = (float*)d_out;
    _Float16* ws  = (_Float16*)d_ws;   // 81,920 bytes of f16 transposed weights

    // 1) transpose+convert weights to f16 (L2-resident, reused by all blocks)
    prep_weights_kernel<<<80, 256, 0, stream>>>(W1, W2, W3, ws);
    // 2) zero output (atomics accumulate into it)
    zero_out_kernel<<<2048, 256, 0, stream>>>(out, out_size / 4);
    // 3) fused edge MLP (WMMA) + async-staged Wigner rotation + scatter
    edge_degree_fused_kernel<<<N_EDGES / EPB, THREADS, 0, stream>>>(
        node_species, dist, senders, receivers, wigner, envelope,
        send_emb, recv_emb, b1, g1, be1, b2, g2, be2, b3, ws, out);
}
